// FFSNN_v2_12120397709824
// MI455X (gfx1250) — matrix operand order repaired and checked by reading, not compiled, not run on hardware
//
#include <hip/hip_runtime.h>

// ---------------------------------------------------------------------------
// FFSNN on MI455X (gfx1250, wave32, WMMA).
//   - 32 blocks x 256 threads (8 waves). Block owns 16 batch rows for all 98
//     timesteps; membrane potentials live in WMMA f32 accumulators (VGPRs).
//   - Spikes (exact {0,1}) stored as f16 in LDS -> directly usable as WMMA A.
//     Previous-step spikes additionally tracked in a per-lane 32-bit bitmask
//     (lane-stable ownership) so the update phase needs NO LDS reads.
//   - Layers 2/3/4: v_wmma_f32_16x16x32_f16 (weights pre-rounded to f16 in ws).
//   - Layer 1 (K=8, fp32 pixels): v_wmma_f32_16x16x4_f32 (exact).
//   - K-loops kept rolled (#pragma unroll 1) + opaque pointer barrier so the
//     compiler cannot hoist the weight streams into (spilled) registers.
// ---------------------------------------------------------------------------

typedef float    v2f  __attribute__((ext_vector_type(2)));
typedef float    v8f  __attribute__((ext_vector_type(8)));
typedef _Float16 v8h  __attribute__((ext_vector_type(8)));
typedef _Float16 v16h __attribute__((ext_vector_type(16)));

#define TSTEPS  98
#define FCD     512
#define HS      520      // LDS row stride in halves: 512 + 8 pad (bank spread)
#define THRESHV 0.3f

// Defeat LICM across the timestep loop: weights must STREAM from L2 each step.
__device__ __forceinline__ const _Float16* opaque(const _Float16* p) {
  asm volatile("" : "+s"(p));
  return p;
}

// A operand (16x32 f16): lane holds row M = lane&15;
// K set = {8*hi+0..7} U {16+8*hi+0..7}  (ISA 7.12.2, 16-bit A 16x32)
__device__ __forceinline__ v16h loadA(const _Float16* __restrict__ row, int k, int hi) {
  v8h a0 = *(const v8h*)(row + k + 8 * hi);
  v8h a1 = *(const v8h*)(row + k + 16 + 8 * hi);
  v16h a;
#pragma unroll
  for (int i = 0; i < 8; ++i) { a[i] = a0[i]; a[8 + i] = a1[i]; }
  return a;
}

// B operand (32x16 f16): lane holds column N = lane&15 (one weight row),
// K set = {8*hi+0..7} U {16+8*hi+0..7} (matches A) -> two b128 loads.
__device__ __forceinline__ v16h loadB(const _Float16* __restrict__ wrow, int k, int hi) {
  v8h b0 = *(const v8h*)(wrow + k + 8 * hi);
  v8h b1 = *(const v8h*)(wrow + k + 16 + 8 * hi);
  v16h b;
#pragma unroll
  for (int i = 0; i < 8; ++i) { b[i] = b0[i]; b[8 + i] = b1[i]; }
  return b;
}

// mem_update_skip_woDecay on one 16x16 C/D tile (lane: N=lane&15, M=v+8*hi).
// Branchless + exact: mk is exactly 0.0 or 1.0, so mk*nw + (1-mk)*mem selects
// exactly (one addend is exact 0). spike = (mem2>0.3) ? mk : 0  == act*mk.
// Previous spike comes from the per-lane bitmask `spk` (bit base+v), not LDS.
__device__ __forceinline__ void update_tile(v8f& mem, const v8f& g, _Float16* sbuf,
                                            unsigned& spk, int base,
                                            int j, float bj,
                                            const int* __restrict__ msk, int t, int hi) {
  const float mk  = (float)msk[j * TSTEPS + t];
  const float nmk = 1.0f - mk;
  unsigned newbits = 0u;
#pragma unroll
  for (int v = 0; v < 8; ++v) {
    const int   mrow = v + 8 * hi;
    const bool  sbit = (spk >> (base + v)) & 1u;
    const float dec  = sbit ? 0.0f : mem[v] * 0.5f;       // mem*DECAY*(1-sprev)
    const float nw   = dec + (g[v] + bj);
    const float m2v  = __builtin_fmaf(mk, nw, nmk * mem[v]);
    mem[v] = m2v;
    const float sp = (m2v > THRESHV) ? mk : 0.0f;
    sbuf[mrow * HS + j] = (_Float16)sp;
    newbits |= (sp != 0.0f ? 1u : 0u) << (base + v);
  }
  spk = (spk & ~(0xFFu << base)) | newbits;
}

// One-shot fp32 -> f16 weight conversion into workspace.
__global__ void ffsnn_cvt_kernel(const float* __restrict__ w2, const float* __restrict__ w3,
                                 const float* __restrict__ w4,
                                 _Float16* __restrict__ w2h, _Float16* __restrict__ w3h,
                                 _Float16* __restrict__ w4h) {
  int idx = blockIdx.x * 256 + threadIdx.x;
  const int NW = 512 * 512;
  if (idx < NW) {
    w2h[idx] = (_Float16)w2[idx];
  } else if (idx < 2 * NW) {
    w3h[idx - NW] = (_Float16)w3[idx - NW];
  } else if (idx < 2 * NW + 16 * 512) {
    int i2 = idx - 2 * NW;
    int c = i2 >> 9, k = i2 & 511;
    w4h[i2] = (_Float16)((c < 10) ? w4[c * 512 + k] : 0.0f);
  }
}

// One rolled K-loop for a 512-K layer: A loaded once per K-step from LDS,
// 4 independent WMMAs (this wave's 4 neuron tiles) for ILP.
__device__ __forceinline__ void gemm512(const _Float16* __restrict__ ar,
                                        const _Float16* __restrict__ wbase,
                                        const int* __restrict__ jn, int hi, v8f c[4]) {
  const _Float16* bw0 = wbase + (size_t)jn[0] * FCD;
  const _Float16* bw1 = wbase + (size_t)jn[1] * FCD;
  const _Float16* bw2 = wbase + (size_t)jn[2] * FCD;
  const _Float16* bw3 = wbase + (size_t)jn[3] * FCD;
#pragma unroll 1
  for (int k = 0; k < FCD; k += 32) {
    v16h a = loadA(ar, k, hi);
    c[0] = __builtin_amdgcn_wmma_f32_16x16x32_f16(false, a, false, loadB(bw0, k, hi), (short)0, c[0], false, false);
    c[1] = __builtin_amdgcn_wmma_f32_16x16x32_f16(false, a, false, loadB(bw1, k, hi), (short)0, c[1], false, false);
    c[2] = __builtin_amdgcn_wmma_f32_16x16x32_f16(false, a, false, loadB(bw2, k, hi), (short)0, c[2], false, false);
    c[3] = __builtin_amdgcn_wmma_f32_16x16x32_f16(false, a, false, loadB(bw3, k, hi), (short)0, c[3], false, false);
  }
}

__global__ void __launch_bounds__(256, 1)
ffsnn_kernel(const float* __restrict__ x,
             const float* __restrict__ w1, const float* __restrict__ b1,
             const float* __restrict__ b2, const float* __restrict__ b3,
             const float* __restrict__ b4,
             const int* __restrict__ pm1, const int* __restrict__ pm2,
             const int* __restrict__ pm3,
             const _Float16* __restrict__ w2h, const _Float16* __restrict__ w3h,
             const _Float16* __restrict__ w4h,
             float* __restrict__ out) {
  __shared__ _Float16 s1[16 * HS];   // layer spikes (f16, {0,1})
  __shared__ _Float16 s2[16 * HS];
  __shared__ _Float16 s3[16 * HS];
  __shared__ float    red[8 * 16 * 16];  // osum partial reduction

  const int tid  = threadIdx.x;
  const int wave = tid >> 5;
  const int lane = tid & 31;
  const int ln   = lane & 15;
  const int hi   = lane >> 4;
  const int rowBase = blockIdx.x * 16;   // 16 batch rows per block

  // Persistent per-wave state: 4 neuron tiles per layer (wave*64 .. wave*64+63)
  v8f mA[4], mB[4], mC[4], osum;
#pragma unroll
  for (int v = 0; v < 8; ++v) osum[v] = 0.0f;
  unsigned spkA = 0u, spkB = 0u, spkC = 0u;   // prev-step spike bits, per lane
  int   jn[4];
  float bi1[4], bi2[4], bi3[4];
#pragma unroll
  for (int nt = 0; nt < 4; ++nt) {
    jn[nt]  = wave * 64 + nt * 16 + ln;           // this lane's neuron column
    bi1[nt] = b1[jn[nt]]; bi2[nt] = b2[jn[nt]]; bi3[nt] = b3[jn[nt]];
#pragma unroll
    for (int v = 0; v < 8; ++v) { mA[nt][v] = 0.0f; mB[nt][v] = 0.0f; mC[nt][v] = 0.0f; }
  }

  // Small t-invariant operands: hoist deliberately (bounded: 16+16 VGPRs).
  v2f w1b0[4], w1b1[4];
#pragma unroll
  for (int nt = 0; nt < 4; ++nt) {
    const int n = jn[nt];
    w1b0[nt].x = w1[n * 8 + 2 * hi];     w1b0[nt].y = w1[n * 8 + 2 * hi + 1];
    w1b1[nt].x = w1[n * 8 + 4 + 2 * hi]; w1b1[nt].y = w1[n * 8 + 4 + 2 * hi + 1];
  }
  v16h w4b0 = loadB(w4h + (size_t)ln * FCD, wave * 64,      hi);
  v16h w4b1 = loadB(w4h + (size_t)ln * FCD, wave * 64 + 32, hi);

  // No LDS pre-zero needed: at t=0 every read entry of s1/s2/s3 is written
  // earlier in the same step behind a barrier, and sprev comes from spk*==0.

#pragma unroll 1
  for (int t = 0; t < TSTEPS; ++t) {
    // faithful time-window slice boundary: start = t*8 if t*8 < 90 else 776
    const int p0 = (t * 8 < TSTEPS - 8) ? t * 8 : (784 - 8);

    // ---------------- layer 1: x[16,8] @ w1^T -> f32 WMMA 16x16x4, K-loop of 2
    {
      const float* xr = x + (size_t)(rowBase + ln) * 784 + p0;
      v2f a0 = { xr[2 * hi],     xr[2 * hi + 1] };      // K 0..3
      v2f a1 = { xr[4 + 2 * hi], xr[4 + 2 * hi + 1] };  // K 4..7
#pragma unroll
      for (int nt = 0; nt < 4; ++nt) {
        v8f c;
#pragma unroll
        for (int v = 0; v < 8; ++v) c[v] = 0.0f;
        c = __builtin_amdgcn_wmma_f32_16x16x4_f32(false, a0, false, w1b0[nt], (short)0, c, false, false);
        c = __builtin_amdgcn_wmma_f32_16x16x4_f32(false, a1, false, w1b1[nt], (short)0, c, false, false);
        update_tile(mA[nt], c, s1, spkA, nt * 8, jn[nt], bi1[nt], pm1, t, hi);
      }
    }
    __syncthreads();

    // ---------------- layer 2: s1[16,512] @ w2^T -> f16 WMMA 16x16x32
    {
      v8f c[4];
#pragma unroll
      for (int nt = 0; nt < 4; ++nt)
#pragma unroll
        for (int v = 0; v < 8; ++v) c[nt][v] = 0.0f;
      gemm512(s1 + ln * HS, opaque(w2h), jn, hi, c);
#pragma unroll
      for (int nt = 0; nt < 4; ++nt)
        update_tile(mB[nt], c[nt], s2, spkB, nt * 8, jn[nt], bi2[nt], pm2, t, hi);
    }
    __syncthreads();

    // ---------------- layer 3: s2 @ w3^T
    {
      v8f c[4];
#pragma unroll
      for (int nt = 0; nt < 4; ++nt)
#pragma unroll
        for (int v = 0; v < 8; ++v) c[nt][v] = 0.0f;
      gemm512(s2 + ln * HS, opaque(w3h), jn, hi, c);
#pragma unroll
      for (int nt = 0; nt < 4; ++nt)
        update_tile(mC[nt], c[nt], s3, spkC, nt * 8, jn[nt], bi3[nt], pm3, t, hi);
    }
    __syncthreads();

    // ---------------- layer 4: osum += s3 @ w4^T  (K split across 8 waves)
    {
      const _Float16* ar = s3 + ln * HS;
      const int kb = wave * 64;
      v16h a0 = loadA(ar, kb,      hi);
      v16h a1 = loadA(ar, kb + 32, hi);
      osum = __builtin_amdgcn_wmma_f32_16x16x32_f16(false, a0, false, w4b0, (short)0, osum, false, false);
      osum = __builtin_amdgcn_wmma_f32_16x16x32_f16(false, a1, false, w4b1, (short)0, osum, false, false);
    }
    // No 4th barrier needed: L3(t+1) writes to s3 sit behind two barriers.
  }

  // ---------------- reduce per-wave osum partials, write out[512,10]
#pragma unroll
  for (int v = 0; v < 8; ++v) red[wave * 256 + (v + 8 * hi) * 16 + ln] = osum[v];
  __syncthreads();
  {
    const int m = tid >> 4, c = tid & 15;
    if (c < 10) {
      float s = 0.0f;
#pragma unroll
      for (int w = 0; w < 8; ++w) s += red[w * 256 + m * 16 + c];
      // sum_t (gemm + b4) / T  ==  (sum_t gemm)/T + b4
      out[(size_t)(rowBase + m) * 10 + c] = s * (1.0f / (float)TSTEPS) + b4[c];
    }
  }
}

extern "C" void kernel_launch(void* const* d_in, const int* in_sizes, int n_in,
                              void* d_out, int out_size, void* d_ws, size_t ws_size,
                              hipStream_t stream) {
  const float* x  = (const float*)d_in[0];
  const float* w1 = (const float*)d_in[1];
  const float* b1 = (const float*)d_in[2];
  const float* w2 = (const float*)d_in[3];
  const float* b2 = (const float*)d_in[4];
  const float* w3 = (const float*)d_in[5];
  const float* b3 = (const float*)d_in[6];
  const float* w4 = (const float*)d_in[7];
  const float* b4 = (const float*)d_in[8];
  const int*   m1 = (const int*)d_in[9];
  const int*   m2 = (const int*)d_in[10];
  const int*   m3 = (const int*)d_in[11];

  _Float16* w2h = (_Float16*)d_ws;            // 512KB
  _Float16* w3h = w2h + 512 * 512;            // 512KB
  _Float16* w4h = w3h + 512 * 512;            // 16KB (16x512, zero-padded rows)

  const int cvtTotal = 2 * 512 * 512 + 16 * 512;
  ffsnn_cvt_kernel<<<(cvtTotal + 255) / 256, 256, 0, stream>>>(w2, w3, w4, w2h, w3h, w4h);

  ffsnn_kernel<<<32, 256, 0, stream>>>(x, w1, b1, b2, b3, b4, m1, m2, m3,
                                       w2h, w3h, w4h, (float*)d_out);
}
